// FractureMambaViT_57509612094220
// MI455X (gfx1250) — matrix operand order repaired and checked by reading, not compiled, not run on hardware
//
#include <hip/hip_runtime.h>

// ---------------- model constants ----------------
#define BATCH   32
#define IMG     224
#define PATCHSZ 16
#define CIN     3
#define DMODEL  192
#define NBLK    4
#define DS      8
#define DKC     4
#define DI      384
#define NPATCH  196
#define LSEQ    197
#define NTOK    (BATCH*LSEQ)     // 6304
#define NPTOK   (BATCH*NPATCH)   // 6272
#define KPATCH  (CIN*PATCHSZ*PATCHSZ) // 768

typedef float  v8f    __attribute__((ext_vector_type(8)));
typedef __bf16 v16bf  __attribute__((ext_vector_type(16)));
typedef __bf16 bf16x8 __attribute__((ext_vector_type(8)));

// ---------------- helpers ----------------
__device__ __forceinline__ unsigned short f2bf_bits(float f) {
  unsigned int u = __builtin_bit_cast(unsigned int, f);
  u += 0x7FFFu + ((u >> 16) & 1u);      // round-to-nearest-even
  return (unsigned short)(u >> 16);
}
__device__ __forceinline__ float siluf(float x) {
  return x / (1.0f + __expf(-x));
}
__device__ __forceinline__ float softplusf(float x) {
  return (x > 20.0f) ? x : log1pf(__expf(x));
}

// A fragment (16x32 bf16): lane m=l&15 row, g=l>>4: K = 8g+j (j<8), 8g+16+j (j>=8)
__device__ __forceinline__ v16bf frag_a(const unsigned short* row, int k, int g) {
  bf16x8 lo = *(const bf16x8*)(row + k + g * 8);
  bf16x8 hi = *(const bf16x8*)(row + k + g * 8 + 16);
  return __builtin_shufflevector(lo, hi, 0,1,2,3,4,5,6,7,8,9,10,11,12,13,14,15);
}
// B fragment (32x16 bf16, stored as W[N][K]): lane col n=l&15, K = 8g+j (j<8), 8g+16+j (j>=8)
__device__ __forceinline__ v16bf frag_b(const unsigned short* row, int k, int g) {
  bf16x8 lo = *(const bf16x8*)(row + k + g * 8);
  bf16x8 hi = *(const bf16x8*)(row + k + g * 8 + 16);
  return __builtin_shufflevector(lo, hi, 0,1,2,3,4,5,6,7,8,9,10,11,12,13,14,15);
}

// ---------------- weight fp32 -> bf16 bits ----------------
__global__ void fm_cvt_bf16(const float* __restrict__ w,
                            unsigned short* __restrict__ o, int n) {
  int i = blockIdx.x * blockDim.x + threadIdx.x;
  if (i < n) o[i] = f2bf_bits(w[i]);
}

// ---------------- im2col (writes bf16 GEMM-A) ----------------
__global__ void fm_im2col(const float* __restrict__ x, unsigned short* __restrict__ out) {
  int idx = blockIdx.x * blockDim.x + threadIdx.x;
  if (idx >= NPTOK * KPATCH) return;
  int k = idx % KPATCH;
  int m = idx / KPATCH;
  int b = m / NPATCH, p = m % NPATCH;
  int ph = p / 14, pw = p % 14;
  int c = k / 256, r = k % 256;
  int py = r / 16, px = r % 16;
  out[idx] = f2bf_bits(
      x[(((size_t)b * CIN + c) * IMG + (ph * 16 + py)) * IMG + (pw * 16 + px)]);
}

// ---------------- WMMA bf16 GEMM: C(MxN,f32) = A(MxK,bf16) * W(NxK,bf16)^T [+bias]
// 256 thr = 8 waves in 2x4; each wave 32x32 (2x2 tiles, 4 wmma per K-step).
__global__ void fm_gemm_wmma(const unsigned short* __restrict__ A,
                             const unsigned short* __restrict__ W,
                             const float* __restrict__ bias,
                             float* __restrict__ C,
                             int M, int N, int K) {
  int wave = threadIdx.x >> 5;
  int lane = threadIdx.x & 31;
  int row0 = blockIdx.x * 64  + (wave & 1) * 32;
  int col0 = blockIdx.y * 128 + (wave >> 1) * 32;
  if (row0 >= M || col0 >= N) return;        // wave-uniform; EXEC all-ones
  int m = lane & 15;
  int g = lane >> 4;
  const unsigned short* Ar0 = A + (size_t)(row0 + m) * K;
  const unsigned short* Ar1 = A + (size_t)(row0 + 16 + m) * K;
  const unsigned short* Br0 = W + (size_t)(col0 + m) * K;
  const unsigned short* Br1 = W + (size_t)(col0 + 16 + m) * K;
  v8f acc00 = {}, acc01 = {}, acc10 = {}, acc11 = {};
  for (int k0 = 0; k0 < K; k0 += 32) {
    if (k0 + 32 < K) {
      __builtin_prefetch(Ar0 + k0 + 32, 0, 1);   // global_prefetch_b8
      __builtin_prefetch(Br0 + k0 + 32, 0, 1);
    }
    v16bf a0 = frag_a(Ar0, k0, g);
    v16bf a1 = frag_a(Ar1, k0, g);
    v16bf b0 = frag_b(Br0, k0, g);
    v16bf b1 = frag_b(Br1, k0, g);
    acc00 = __builtin_amdgcn_wmma_f32_16x16x32_bf16(false, a0, false, b0, (short)0, acc00, false, false);
    acc01 = __builtin_amdgcn_wmma_f32_16x16x32_bf16(false, a0, false, b1, (short)0, acc01, false, false);
    acc10 = __builtin_amdgcn_wmma_f32_16x16x32_bf16(false, a1, false, b0, (short)0, acc10, false, false);
    acc11 = __builtin_amdgcn_wmma_f32_16x16x32_bf16(false, a1, false, b1, (short)0, acc11, false, false);
  }
  int n0 = col0 + m, n1 = col0 + 16 + m;
  float bv0 = bias ? bias[n0] : 0.0f;
  float bv1 = bias ? bias[n1] : 0.0f;
#pragma unroll
  for (int p = 0; p < 8; ++p) {
    int r0 = row0 + p + 8 * g;          // D-matrix row = p + 8g
    int r1 = r0 + 16;
    C[(size_t)r0 * N + n0] = acc00[p] + bv0;
    C[(size_t)r0 * N + n1] = acc01[p] + bv1;
    C[(size_t)r1 * N + n0] = acc10[p] + bv0;
    C[(size_t)r1 * N + n1] = acc11[p] + bv1;
  }
}

// ---------------- LayerNorm over D=192, one wave32 per token ----------------
// Writes f32 out; optionally also bf16 (for GEMM-A staging).
__global__ void fm_layernorm(const float* __restrict__ x,
                             const float* __restrict__ gam,
                             const float* __restrict__ bet,
                             float* __restrict__ out,
                             unsigned short* __restrict__ obf,
                             int ntok) {
  int tok  = blockIdx.x * (blockDim.x >> 5) + (threadIdx.x >> 5);
  int lane = threadIdx.x & 31;
  if (tok >= ntok) return;
  const float* xr = x + (size_t)tok * DMODEL;
  float v[6], s = 0.0f;
#pragma unroll
  for (int i = 0; i < 6; ++i) { v[i] = xr[lane + 32 * i]; s += v[i]; }
#pragma unroll
  for (int o = 16; o > 0; o >>= 1) s += __shfl_xor(s, o, 32);
  float mean = s * (1.0f / DMODEL);
  float q = 0.0f;
#pragma unroll
  for (int i = 0; i < 6; ++i) { float d = v[i] - mean; q += d * d; }
#pragma unroll
  for (int o = 16; o > 0; o >>= 1) q += __shfl_xor(q, o, 32);
  float rstd = rsqrtf(q * (1.0f / DMODEL) + 1e-5f);
#pragma unroll
  for (int i = 0; i < 6; ++i) {
    int d = lane + 32 * i;
    float y = (v[i] - mean) * rstd * gam[d] + bet[d];
    out[(size_t)tok * DMODEL + d] = y;
    if (obf) obf[(size_t)tok * DMODEL + d] = f2bf_bits(y);
  }
}

// ---------------- assemble h = [cls; pe_ln] + pos ----------------
__global__ void fm_build_h(const float* __restrict__ peln,
                           const float* __restrict__ cls,
                           const float* __restrict__ pos,
                           float* __restrict__ h) {
  int idx = blockIdx.x * blockDim.x + threadIdx.x;
  if (idx >= NTOK * DMODEL) return;
  int d = idx % DMODEL;
  int l = (idx / DMODEL) % LSEQ;
  int b = idx / (DMODEL * LSEQ);
  if (l == 0)
    h[idx] = cls[d] + pos[d];
  else
    h[idx] = peln[((size_t)(b * NPATCH + (l - 1))) * DMODEL + d] + pos[l * DMODEL + d];
}

// ---------------- causal depthwise conv (DK=4) + SiLU ----------------
__global__ void fm_conv_silu(const float* __restrict__ xz,   // (B,L,768)
                             const float* __restrict__ cw,
                             const float* __restrict__ cb,
                             float* __restrict__ xc,         // (B,L,384) t-order
                             int reverse) {
  int idx = blockIdx.x * blockDim.x + threadIdx.x;
  if (idx >= NTOK * DI) return;
  int c = idx % DI;
  int t = (idx / DI) % LSEQ;
  int b = idx / (DI * LSEQ);
  float acc = cb[c];
#pragma unroll
  for (int k = 0; k < DKC; ++k) {
    int l = t - (DKC - 1) + k;
    if (l >= 0) {
      int lm = reverse ? (LSEQ - 1 - l) : l;
      acc += cw[c * DKC + k] * xz[((size_t)(b * LSEQ + lm)) * (2 * DI) + c];
    }
  }
  xc[idx] = siluf(acc);
}

// ---------------- s = xc @ x_w^T  (N = 17) ----------------
__global__ void fm_sproj(const float* __restrict__ xc,
                         const float* __restrict__ xw,
                         float* __restrict__ sb) {
  int idx = blockIdx.x * blockDim.x + threadIdx.x;
  if (idx >= NTOK * 17) return;
  int j = idx % 17, tok = idx / 17;
  const float* xr = xc + (size_t)tok * DI;
  const float* wr = xw + j * DI;
  float s = 0.0f;
  for (int k = 0; k < DI; ++k) s += xr[k] * wr[k];
  sb[idx] = s;
}

// ---------------- selective scan: one thread per (b,c) ----------------
__global__ void fm_scan(const float* __restrict__ xc,
                        const float* __restrict__ sb,
                        const float* __restrict__ A_log,
                        const float* __restrict__ dt_w,
                        const float* __restrict__ dt_b,
                        const float* __restrict__ Dp,
                        float* __restrict__ y) {
  int c = blockIdx.x * blockDim.x + threadIdx.x;
  int b = blockIdx.y;
  if (c >= DI) return;
  float Ac[DS];
#pragma unroll
  for (int s = 0; s < DS; ++s) Ac[s] = -__expf(A_log[c * DS + s]);
  const float dtw = dt_w[c], dtb = dt_b[c], Dc = Dp[c];
  float hs[DS];
#pragma unroll
  for (int s = 0; s < DS; ++s) hs[s] = 0.0f;
  for (int t = 0; t < LSEQ; ++t) {
    size_t tok = (size_t)b * LSEQ + t;
    float u   = xc[tok * DI + c];
    float dt  = softplusf(sb[tok * 17 + 16] * dtw + dtb);
    float acc = 0.0f;
#pragma unroll
    for (int s = 0; s < DS; ++s) {
      float Bs = sb[tok * 17 + s];
      float Cs = sb[tok * 17 + 8 + s];
      float dA = __expf(dt * Ac[s]);
      hs[s] = dA * hs[s] + dt * Bs * u;
      acc  += hs[s] * Cs;
    }
    y[tok * DI + c] = acc + u * Dc;
  }
}

// ---------------- gate: ybf = bf16( y * silu(z[map]) ) ----------------
__global__ void fm_gate(const float* __restrict__ y,
                        const float* __restrict__ xz,
                        unsigned short* __restrict__ ybf,
                        int reverse) {
  int idx = blockIdx.x * blockDim.x + threadIdx.x;
  if (idx >= NTOK * DI) return;
  int c = idx % DI;
  int t = (idx / DI) % LSEQ;
  int b = idx / (DI * LSEQ);
  int lm = reverse ? (LSEQ - 1 - t) : t;
  float z = xz[((size_t)(b * LSEQ + lm)) * (2 * DI) + DI + c];
  ybf[idx] = f2bf_bits(y[idx] * siluf(z));
}

// ---------------- catbf[b,l,dirOff+d] = bf16( h + o[map] ) ----------------
__global__ void fm_accum_cat(const float* __restrict__ h,
                             const float* __restrict__ o,
                             unsigned short* __restrict__ cat,
                             int dirOff, int reverse) {
  int idx = blockIdx.x * blockDim.x + threadIdx.x;
  if (idx >= NTOK * DMODEL) return;
  int d = idx % DMODEL;
  int l = (idx / DMODEL) % LSEQ;
  int b = idx / (DMODEL * LSEQ);
  int tm = reverse ? (LSEQ - 1 - l) : l;
  cat[((size_t)(b * LSEQ + l)) * (2 * DMODEL) + dirOff + d] =
      f2bf_bits(h[idx] + o[((size_t)(b * LSEQ + tm)) * DMODEL + d]);
}

// ---------------- final output packing ----------------
__global__ void fm_out_final(const float* __restrict__ hln, float* __restrict__ out) {
  int idx = blockIdx.x * blockDim.x + threadIdx.x;
  if (idx >= NTOK * DMODEL) return;
  int d = idx % DMODEL;
  int l = (idx / DMODEL) % LSEQ;
  int b = idx / (DMODEL * LSEQ);
  if (l == 0)
    out[(size_t)b * DMODEL + d] = hln[idx];
  else
    out[(size_t)BATCH * DMODEL + ((size_t)(b * NPATCH + (l - 1))) * DMODEL + d] = hln[idx];
}

// =======================================================================
extern "C" void kernel_launch(void* const* d_in, const int* in_sizes, int n_in,
                              void* d_out, int out_size, void* d_ws, size_t ws_size,
                              hipStream_t stream) {
  (void)in_sizes; (void)n_in; (void)out_size; (void)ws_size;
  const float* X       = (const float*)d_in[0];
  const float* patch_w = (const float*)d_in[1];
  const float* patch_b = (const float*)d_in[2];
  const float* pln_g   = (const float*)d_in[3];
  const float* pln_b   = (const float*)d_in[4];
  const float* cls     = (const float*)d_in[5];
  const float* pos     = (const float*)d_in[6];
  auto BLK  = [&](int i) { return 7 + 26 * i; };
  const int F_LNG=0, F_LNB=1, F_INW=2, F_CONVW=3, F_CONVB=4, F_XW=5,
            F_ALOG=6, F_D=7, F_DTW=8, F_DTB=9, F_OUTW=10;
  const float* fin_g = (const float*)d_in[7 + 26 * NBLK];
  const float* fin_b = (const float*)d_in[7 + 26 * NBLK + 1];

  // -------- workspace layout: f32 region, then 64B-aligned bf16 region --------
  float* f = (float*)d_ws;
  size_t o = 0;
  float* big0 = f + o; o += (size_t)NTOK * 768;      // xz (f32)
  float* hbuf = f + o; o += (size_t)NTOK * DMODEL;
  float* xn   = f + o; o += (size_t)NTOK * DMODEL;
  float* xcb  = f + o; o += (size_t)NTOK * DI;
  float* yb   = f + o; o += (size_t)NTOK * DI;       // scan out / pe_raw scratch
  float* sb   = f + o; o += (size_t)NTOK * 17;
  o = (o + 15) & ~(size_t)15;                        // 64B align for bf16 region
  unsigned short* u = (unsigned short*)(f + o);
  size_t uo = 0;
  unsigned short* imcol_bf = u + uo; uo += (size_t)NPTOK * KPATCH;
  unsigned short* xnbf     = u + uo; uo += (size_t)NTOK * DMODEL;
  unsigned short* ybf      = u + uo; uo += (size_t)NTOK * DI;
  unsigned short* catbf    = u + uo; uo += (size_t)NTOK * 2 * DMODEL;
  unsigned short* wbf      = u + uo;

  size_t wboff = 0;
  auto CVT = [&](const float* src, size_t n) -> const unsigned short* {
    unsigned short* dst = wbf + wboff; wboff += n;
    fm_cvt_bf16<<<(int)((n + 255) / 256), 256, 0, stream>>>(src, dst, (int)n);
    return dst;
  };
  const unsigned short* wb_patch = CVT(patch_w, (size_t)DMODEL * KPATCH);
  const unsigned short* wb_in[NBLK][2];
  const unsigned short* wb_out[NBLK][2];
  const unsigned short* wb_fuse[NBLK];
  for (int i = 0; i < NBLK; ++i) {
    for (int dir = 0; dir < 2; ++dir) {
      int base = BLK(i) + dir * 11;
      wb_in[i][dir]  = CVT((const float*)d_in[base + F_INW],  (size_t)(2 * DI) * DMODEL);
      wb_out[i][dir] = CVT((const float*)d_in[base + F_OUTW], (size_t)DMODEL * DI);
    }
    wb_fuse[i] = CVT((const float*)d_in[BLK(i) + 22], (size_t)DMODEL * (2 * DMODEL));
  }

  auto GEMM = [&](const unsigned short* A, const unsigned short* W, const float* bias,
                  float* C, int M, int N, int K) {
    dim3 grid((M + 63) / 64, (N + 127) / 128);
    fm_gemm_wmma<<<grid, 256, 0, stream>>>(A, W, bias, C, M, N, K);
  };
  auto LN = [&](const float* in, const float* g, const float* b,
                float* out, unsigned short* obf, int nt) {
    fm_layernorm<<<(nt + 7) / 8, 256, 0, stream>>>(in, g, b, out, obf, nt);
  };
  const int EW = 256;
  const int NE_D  = (NTOK * DMODEL + EW - 1) / EW;
  const int NE_DI = (NTOK * DI + EW - 1) / EW;

  // -------- patch embedding --------
  fm_im2col<<<(NPTOK * KPATCH + EW - 1) / EW, EW, 0, stream>>>(X, imcol_bf);
  GEMM(imcol_bf, wb_patch, patch_b, yb, NPTOK, DMODEL, KPATCH);
  LN(yb, pln_g, pln_b, xn, nullptr, NPTOK);
  fm_build_h<<<NE_D, EW, 0, stream>>>(xn, cls, pos, hbuf);

  // -------- bidirectional mamba blocks --------
  for (int i = 0; i < NBLK; ++i) {
    for (int dir = 0; dir < 2; ++dir) {
      int base = BLK(i) + dir * 11;
      LN(hbuf, (const float*)d_in[base + F_LNG], (const float*)d_in[base + F_LNB],
         xn, xnbf, NTOK);
      GEMM(xnbf, wb_in[i][dir], nullptr, big0, NTOK, 2 * DI, DMODEL);      // xz
      fm_conv_silu<<<NE_DI, EW, 0, stream>>>(big0,
          (const float*)d_in[base + F_CONVW], (const float*)d_in[base + F_CONVB],
          xcb, dir);
      fm_sproj<<<(NTOK * 17 + EW - 1) / EW, EW, 0, stream>>>(xcb,
          (const float*)d_in[base + F_XW], sb);
      fm_scan<<<dim3(DI / 128, BATCH), 128, 0, stream>>>(xcb, sb,
          (const float*)d_in[base + F_ALOG], (const float*)d_in[base + F_DTW],
          (const float*)d_in[base + F_DTB],  (const float*)d_in[base + F_D], yb);
      fm_gate<<<NE_DI, EW, 0, stream>>>(yb, big0, ybf, dir);
      GEMM(ybf, wb_out[i][dir], nullptr, xn, NTOK, DMODEL, DI);            // o
      fm_accum_cat<<<NE_D, EW, 0, stream>>>(hbuf, xn, catbf, dir * DMODEL, dir);
    }
    GEMM(catbf, wb_fuse[i], (const float*)d_in[BLK(i) + 23], xn, NTOK, DMODEL, 2 * DMODEL);
    LN(xn, (const float*)d_in[BLK(i) + 24], (const float*)d_in[BLK(i) + 25],
       hbuf, nullptr, NTOK);
  }

  // -------- final LN + output packing --------
  LN(hbuf, fin_g, fin_b, xn, nullptr, NTOK);
  fm_out_final<<<NE_D, EW, 0, stream>>>(xn, (float*)d_out);
}
